// causalAttention_v3_76278619177479
// MI455X (gfx1250) — matrix operand order repaired and checked by reading, not compiled, not run on hardware
//
#include <hip/hip_runtime.h>
#include <hip/hip_bf16.h>
#include <stdint.h>

#define BATCH 4
#define CTX   2048
#define DIM   1024
#define INV_SQRT_DK 0.03125f   // 1/sqrt(1024)

typedef __attribute__((ext_vector_type(16))) __bf16 bf16x16;
typedef __attribute__((ext_vector_type(8)))  float  f32x8;

union Frag16 {
    bf16x16       v;
    uint4         q[2];
    unsigned int  u[8];
};

__device__ __forceinline__ unsigned short f2bf(float f) {
    union { float f; unsigned int u; } x; x.f = f;
    unsigned int r = x.u + 0x7FFFu + ((x.u >> 16) & 1u);   // round-to-nearest-even
    return (unsigned short)(r >> 16);
}
__device__ __forceinline__ float bf2f(unsigned short h) {
    union { unsigned int u; float f; } x; x.u = ((unsigned int)h) << 16;
    return x.f;
}
__device__ __forceinline__ unsigned int pack2(float lo, float hi) {
    return ((unsigned int)f2bf(hi) << 16) | (unsigned int)f2bf(lo);
}

// A fragment: lane = half*16 + m; elements 0..7 = K(d0+half*8+j), 8..15 = K(d0+16+half*8+j)
__device__ __forceinline__ Frag16 ldA(const unsigned short* row, int d0, int half) {
    Frag16 F;
    F.q[0] = *(const uint4*)(row + d0 + half * 8);
    F.q[1] = *(const uint4*)(row + d0 + 16 + half * 8);
    return F;
}
// B fragment: lane = half*16 + n; elements 0..7 = K(d0+half*8+j), 8..15 = K(d0+16+half*8+j)
__device__ __forceinline__ Frag16 ldB(const unsigned short* row, int d0, int half) {
    Frag16 F;
    F.q[0] = *(const uint4*)(row + d0 + half * 8);
    F.q[1] = *(const uint4*)(row + d0 + 16 + half * 8);
    return F;
}

#define WMMA_BF16(A, B, C) \
    __builtin_amdgcn_wmma_f32_16x16x32_bf16(false, (A).v, false, (B).v, (short)0, (C), false, false)

// ---------------------------------------------------------------------------
// Kernel 0a: zero the per-(batch,key) denominator accumulator
// ---------------------------------------------------------------------------
__global__ void caus_init_denom(float* __restrict__ denom) {
    int i = blockIdx.x * blockDim.x + threadIdx.x;
    if (i < BATCH * CTX) denom[i] = 0.0f;
}

// ---------------------------------------------------------------------------
// Kernel 0b: f32 -> bf16 bulk conversion (x and the weight matrices), done
// once so the GEMM inner loops are pure bf16 loads + WMMA.
// ---------------------------------------------------------------------------
__global__ __launch_bounds__(256) void caus_cvt_bf16(
    const float* __restrict__ src, unsigned int* __restrict__ dst, int n4)
{
    int i = blockIdx.x * blockDim.x + threadIdx.x;
    if (i >= n4) return;
    float4 f = ((const float4*)src)[i];
    uint2 o;
    o.x = pack2(f.x, f.y);
    o.y = pack2(f.z, f.w);
    ((uint2*)dst)[i] = o;
}

// ---------------------------------------------------------------------------
// Kernel 1: QKV projections, bf16 GEMM, 64x64 register-blocked per wave:
// 4 A-frags x 4 B-frags -> 16 WMMA per 16 loads. Q,K stored bf16 row-major;
// V stored TRANSPOSED as Vt[b][d_out][t] for contiguous B loads in kernel 4.
// ---------------------------------------------------------------------------
__global__ __launch_bounds__(256) void caus_qkv_proj(
    const unsigned short* __restrict__ xb, const unsigned short* __restrict__ Wb,
    unsigned short* __restrict__ Qb, unsigned short* __restrict__ Kb,
    unsigned short* __restrict__ Vt)
{
    const int lane = threadIdx.x & 31;
    const int wv   = threadIdx.x >> 5;
    const int tile = blockIdx.x * 8 + wv;        // 3 * 128 * 16 tiles
    const int mat  = tile / (128 * 16);          // 0=Q 1=K 2=V
    const int rem  = tile % (128 * 16);
    const int rT   = rem / 16;                   // 64-row block over 8192 rows
    const int cT   = rem % 16;                   // 64-col block over 1024 cols
    const int ln   = lane & 15;
    const int half = lane >> 4;
    const int row0 = rT * 64;
    const int col0 = cT * 64;

    const unsigned short* xrow[4];
    const unsigned short* wrow[4];
    const unsigned short* W = Wb + (size_t)mat * DIM * DIM;
    #pragma unroll
    for (int i = 0; i < 4; ++i) {
        xrow[i] = xb + (size_t)(row0 + i * 16 + ln) * DIM;
        wrow[i] = W  + (size_t)(col0 + i * 16 + ln) * DIM;
    }

    f32x8 acc[16] = {};
    #pragma unroll 1
    for (int d0 = 0; d0 < DIM; d0 += 32) {
        Frag16 A[4], B[4];
        #pragma unroll
        for (int i = 0; i < 4; ++i) A[i] = ldA(xrow[i], d0, half);
        #pragma unroll
        for (int j = 0; j < 4; ++j) B[j] = ldB(wrow[j], d0, half);
        #pragma unroll
        for (int i = 0; i < 4; ++i)
            #pragma unroll
            for (int j = 0; j < 4; ++j)
                acc[i * 4 + j] = WMMA_BF16(A[i], B[j], acc[i * 4 + j]);
    }

    if (mat == 2) {
        #pragma unroll
        for (int i = 0; i < 4; ++i)
            #pragma unroll
            for (int r = 0; r < 8; ++r) {
                int row = row0 + i * 16 + half * 8 + r;   // global row over 8192
                int b   = row >> 11;
                int t   = row & (CTX - 1);
                #pragma unroll
                for (int j = 0; j < 4; ++j) {
                    int d = col0 + j * 16 + ln;
                    Vt[((size_t)(b * DIM + d) << 11) + t] = f2bf(acc[i * 4 + j][r]);
                }
            }
    } else {
        unsigned short* O = (mat == 0) ? Qb : Kb;
        #pragma unroll
        for (int i = 0; i < 4; ++i)
            #pragma unroll
            for (int r = 0; r < 8; ++r) {
                int row = row0 + i * 16 + half * 8 + r;
                #pragma unroll
                for (int j = 0; j < 4; ++j)
                    O[(size_t)row * DIM + col0 + j * 16 + ln] = f2bf(acc[i * 4 + j][r]);
            }
    }
}

// ---------------------------------------------------------------------------
// Kernel 2: S = Q @ K^T per 64x64 block (16 WMMA / 16 loads per K-step),
// causal mask, E = exp(S/sqrt(dk)) stored bf16, per-key column sums
// accumulated atomically (reference softmax is over the QUERY axis).
// Fully-masked blocks (kT64 > qT64) are never read downstream -> early out.
// ---------------------------------------------------------------------------
__global__ __launch_bounds__(256) void caus_scores(
    const unsigned short* __restrict__ Qb, const unsigned short* __restrict__ Kb,
    unsigned short* __restrict__ E, float* __restrict__ denom)
{
    const int lane = threadIdx.x & 31;
    const int wv   = threadIdx.x >> 5;
    const int tile = blockIdx.x * 8 + wv;        // 4 * 32 * 32 blocks
    const int b    = tile / (32 * 32);
    const int r2   = tile % (32 * 32);
    const int qT   = r2 / 32;
    const int kT   = r2 % 32;
    if (kT > qT) return;                          // fully masked, never read

    const int ln   = lane & 15;
    const int half = lane >> 4;
    const int q0   = qT * 64;
    const int k0   = kT * 64;
    unsigned short* Eb = E + ((size_t)b << 22);   // 2048*2048 per batch

    const unsigned short* qrow[4];
    const unsigned short* krow[4];
    #pragma unroll
    for (int i = 0; i < 4; ++i) {
        qrow[i] = Qb + (size_t)(b * CTX + q0 + i * 16 + ln) * DIM;
        krow[i] = Kb + (size_t)(b * CTX + k0 + i * 16 + ln) * DIM;
    }

    f32x8 acc[16] = {};
    #pragma unroll 1
    for (int d0 = 0; d0 < DIM; d0 += 32) {
        Frag16 A[4], B[4];
        #pragma unroll
        for (int i = 0; i < 4; ++i) A[i] = ldA(qrow[i], d0, half);
        #pragma unroll
        for (int j = 0; j < 4; ++j) B[j] = ldB(krow[j], d0, half);
        #pragma unroll
        for (int i = 0; i < 4; ++i)
            #pragma unroll
            for (int j = 0; j < 4; ++j)
                acc[i * 4 + j] = WMMA_BF16(A[i], B[j], acc[i * 4 + j]);
    }

    float colsum[4] = {0.0f, 0.0f, 0.0f, 0.0f};
    #pragma unroll
    for (int i = 0; i < 4; ++i)
        #pragma unroll
        for (int r = 0; r < 8; ++r) {
            int q = q0 + i * 16 + half * 8 + r;
            #pragma unroll
            for (int j = 0; j < 4; ++j) {
                int   k = k0 + j * 16 + ln;
                float e = (q >= k) ? __expf(acc[i * 4 + j][r] * INV_SQRT_DK) : 0.0f;
                colsum[j] += e;
                Eb[((size_t)q << 11) + k] = f2bf(e);
            }
        }
    #pragma unroll
    for (int j = 0; j < 4; ++j) {
        float s = colsum[j] + __shfl_xor(colsum[j], 16, 32); // fold half-lanes
        if (lane < 16) atomicAdd(&denom[b * CTX + k0 + j * 16 + lane], s);
    }
}

// ---------------------------------------------------------------------------
// Kernel 3: V'[k][d] = V[k][d] / denom[k]  (on transposed Vt, 4 elems/thread)
// ---------------------------------------------------------------------------
__global__ __launch_bounds__(256) void caus_scale_v(
    unsigned short* __restrict__ Vt, const float* __restrict__ denom)
{
    int idx4 = blockIdx.x * blockDim.x + threadIdx.x;   // 2,097,152 threads
    int base = idx4 * 4;                                 // element index
    int t0   = base & (CTX - 1);
    int b    = base >> 21;                               // (base>>11)>>10
    const float* dn = denom + b * CTX + t0;
    uint2 vv = ((const uint2*)Vt)[idx4];
    unsigned short h0 = (unsigned short)(vv.x & 0xFFFF), h1 = (unsigned short)(vv.x >> 16);
    unsigned short h2 = (unsigned short)(vv.y & 0xFFFF), h3 = (unsigned short)(vv.y >> 16);
    uint2 o;
    o.x = pack2(bf2f(h0) / dn[0], bf2f(h1) / dn[1]);
    o.y = pack2(bf2f(h2) / dn[2], bf2f(h3) / dn[3]);
    ((uint2*)Vt)[idx4] = o;
}

// ---------------------------------------------------------------------------
// Kernel 4: out = E @ V', 64x64 register-blocked.  A = E rows (contiguous
// over k); B[k][d] = Vt[d][k] so B loads are contiguous Vt rows.  Causal
// bound: nk = 2*qT64+2 chunks of K=32 reaches exactly k = q0+63; blocks
// above the diagonal are never touched.
// ---------------------------------------------------------------------------
__global__ __launch_bounds__(256) void caus_out_gemm(
    const unsigned short* __restrict__ E, const unsigned short* __restrict__ Vt,
    float* __restrict__ out)
{
    const int lane = threadIdx.x & 31;
    const int wv   = threadIdx.x >> 5;
    const int tile = blockIdx.x * 8 + wv;        // 4 * 32 * 16 blocks
    const int b    = tile / (32 * 16);
    const int r2   = tile % (32 * 16);
    const int qT   = r2 / 16;
    const int dT   = r2 % 16;
    const int ln   = lane & 15;
    const int half = lane >> 4;
    const int q0   = qT * 64;
    const int d0c  = dT * 64;

    const unsigned short* Erow[4];
    const unsigned short* Vrow[4];
    #pragma unroll
    for (int i = 0; i < 4; ++i) {
        Erow[i] = E  + ((size_t)b << 22) + (size_t)(q0 + i * 16 + ln) * CTX;
        Vrow[i] = Vt + ((size_t)(b * DIM + d0c + i * 16 + ln) << 11);
    }

    f32x8 acc[16] = {};
    const int nk = qT * 2 + 2;                    // chunks of K=32, k <= q0+63
    #pragma unroll 1
    for (int c = 0; c < nk; ++c) {
        int kc = c * 32;
        __builtin_prefetch(Erow[0] + kc + 64, 0, 1);
        Frag16 A[4], B[4];
        #pragma unroll
        for (int i = 0; i < 4; ++i) A[i] = ldA(Erow[i], kc, half);
        #pragma unroll
        for (int j = 0; j < 4; ++j) B[j] = ldB(Vrow[j], kc, half);
        #pragma unroll
        for (int i = 0; i < 4; ++i)
            #pragma unroll
            for (int j = 0; j < 4; ++j)
                acc[i * 4 + j] = WMMA_BF16(A[i], B[j], acc[i * 4 + j]);
    }

    #pragma unroll
    for (int i = 0; i < 4; ++i)
        #pragma unroll
        for (int r = 0; r < 8; ++r) {
            int q = q0 + i * 16 + half * 8 + r;
            #pragma unroll
            for (int j = 0; j < 4; ++j)
                out[((size_t)(b * CTX + q)) * DIM + d0c + j * 16 + ln] = acc[i * 4 + j][r];
        }
}

// ---------------------------------------------------------------------------
// Workspace layout (80 MB + 32 KB). xb/Wb are dead after the projection, so
// E (32 MB) overlays the [0, 32 MB) region.
//   xb : [ 0 MB, 16 MB)   bf16 x            (proj only)
//   Wb : [16 MB, 22 MB)   bf16 Wq|Wk|Wv     (proj only)
//   E  : [ 0 MB, 32 MB)   bf16 exp-scores   (from caus_scores onward)
//   Qb : [32 MB, 48 MB)   bf16 Q
//   Kb : [48 MB, 64 MB)   bf16 K
//   Vt : [64 MB, 80 MB)   bf16 V transposed [b][d][t]
//   dn : [80 MB, +32 KB)  f32 column sums
// ---------------------------------------------------------------------------
extern "C" void kernel_launch(void* const* d_in, const int* in_sizes, int n_in,
                              void* d_out, int out_size, void* d_ws, size_t ws_size,
                              hipStream_t stream) {
    const float* x  = (const float*)d_in[0];
    const float* Wq = (const float*)d_in[1];
    const float* Wk = (const float*)d_in[2];
    const float* Wv = (const float*)d_in[3];
    float* out = (float*)d_out;

    const size_t MB = 1024 * 1024;
    char* ws = (char*)d_ws;
    unsigned short* xb = (unsigned short*)(ws);
    unsigned short* Wb = (unsigned short*)(ws + 16 * MB);
    unsigned short* E  = (unsigned short*)(ws);            // overlays xb/Wb
    unsigned short* Qb = (unsigned short*)(ws + 32 * MB);
    unsigned short* Kb = (unsigned short*)(ws + 48 * MB);
    unsigned short* Vt = (unsigned short*)(ws + 64 * MB);
    float*          dn = (float*)        (ws + 80 * MB);

    const int n4x = BATCH * CTX * DIM / 4;    // 2,097,152
    const int n4w = DIM * DIM / 4;            //   262,144

    caus_init_denom<<<(BATCH * CTX + 255) / 256, 256, 0, stream>>>(dn);
    caus_cvt_bf16<<<(n4x + 255) / 256, 256, 0, stream>>>(x,  (unsigned int*)xb, n4x);
    caus_cvt_bf16<<<(n4w + 255) / 256, 256, 0, stream>>>(Wq, (unsigned int*)(Wb),                 n4w);
    caus_cvt_bf16<<<(n4w + 255) / 256, 256, 0, stream>>>(Wk, (unsigned int*)(Wb + DIM * DIM),     n4w);
    caus_cvt_bf16<<<(n4w + 255) / 256, 256, 0, stream>>>(Wv, (unsigned int*)(Wb + 2 * DIM * DIM), n4w);

    caus_qkv_proj<<<768, 256, 0, stream>>>(xb, Wb, Qb, Kb, Vt);
    caus_scores  <<<512, 256, 0, stream>>>(Qb, Kb, E, dn);
    caus_scale_v <<<8192, 256, 0, stream>>>(Vt, dn);
    caus_out_gemm<<<256, 256, 0, stream>>>(E, Vt, out);
}
